// LSTM_15418932592859
// MI455X (gfx1250) — matrix operand order repaired and checked by reading, not compiled, not run on hardware
//
#include <hip/hip_runtime.h>
#include <hip/hip_fp16.h>

// ---------------------------------------------------------------------------
// Persistent fused 2-layer LSTM for MI455X (gfx1250, wave32, WMMA).
//
//   gates[32,2048] = [x_t | h_{t-1}] (32x1024, f16) @ W_cat (1024x2048, f16)
//   accumulated in f32 via v_wmma_f32_16x16x32_f16, then elementwise LSTM
//   cell update in f32. Layer1 is pipelined one step behind layer0 so both
//   layers' GEMMs run in the same grid-sync round (2049 rounds, 2 syncs each).
//
//   h-state staging uses GLOBAL_LOAD_ASYNC_TO_LDS_B128 (ASYNCcnt) when the
//   toolchain exposes the builtin; otherwise a plain copy.
//
// WMMA fragment lane layouts (ISA 7.12.2), flattened as 16 halves per lane:
//   A (16x32 MxK):  lane = m + 16*((k>>3)&1),  idx = (k&7) | ((k>>4)<<3)
//   B (32x16 KxN):  lane = n + 16*((k>>3)&1),  idx = (k&7) | ((k>>4)<<3)
//   C (16x16 MxN):  lane = n + 16*((m>>3)&1),  vgpr = m & 7
// ---------------------------------------------------------------------------

typedef __attribute__((ext_vector_type(16))) _Float16 v16h;
typedef __attribute__((ext_vector_type(4)))  _Float16 v4h;
typedef __attribute__((ext_vector_type(8)))  float    v8f;
typedef __attribute__((ext_vector_type(4)))  int      v4i;

#define TT   2048
#define NWG  32      // 16 workgroups per layer

#if defined(__AMDGCN__) && __has_builtin(__builtin_amdgcn_global_load_async_to_lds_b128)
#define HAS_ASYNC_LDS 1
#define AS1 __attribute__((address_space(1)))
#define AS3 __attribute__((address_space(3)))
#else
#define HAS_ASYNC_LDS 0
#endif

// Workspace layout (bytes, all 256-aligned):
//   [0,      256)      sync counter
//   [256,    +131072)  c state  f32 [2][32*512]
//   [131328, +65536)   hfrag    f16 [2][2mt][16kt][512]  (A-swizzled h state)
//   [196864, +32768)   hxfer    f16 [2mt][16kt][512]     (layer0 h -> layer1 x)
//   [229632, +16384)   biasCat  f32 [2][2048]            (b_ih + b_hh)
//   [246016, +524288)  gbuf     f32 [2][256 tiles][256]  (gate accumulators)
//   [508160, +8388608) Wpack    f16 [2][32kt][128nt][512](B-swizzled W_cat)
// total 8,896,768 bytes

__device__ __forceinline__ float fsig(float x){ return 1.f/(1.f + __expf(-x)); }
__device__ __forceinline__ float ftanh(float x){
  float t = __expf(-2.f * fabsf(x));
  float r = (1.f - t) / (1.f + t);
  return copysignf(r, x);
}

__device__ __forceinline__ void async_copy16(const void* g, void* l){
#if HAS_ASYNC_LDS
  __builtin_amdgcn_global_load_async_to_lds_b128(
      (AS1 v4i*)(v4i*)(void*)g, (AS3 v4i*)(v4i*)l, 0, 0);
#else
  (void)g; (void)l;
#endif
}

__device__ __forceinline__ void async_wait(){
#if HAS_ASYNC_LDS
#if __has_builtin(__builtin_amdgcn_s_wait_asynccnt)
  __builtin_amdgcn_s_wait_asynccnt(0);
#else
  asm volatile("s_wait_asynccnt 0x0" ::: "memory");
#endif
#endif
}

__device__ __forceinline__ void grid_sync(unsigned* ctr, unsigned target){
  __threadfence();
  __syncthreads();
  if (threadIdx.x == 0){
    __hip_atomic_fetch_add(ctr, 1u, __ATOMIC_ACQ_REL, __HIP_MEMORY_SCOPE_AGENT);
    while (__hip_atomic_load(ctr, __ATOMIC_ACQUIRE, __HIP_MEMORY_SCOPE_AGENT) < target)
      __builtin_amdgcn_s_sleep(2);
  }
  __syncthreads();
}

extern "C" __global__ void lstm_init_kernel(unsigned* ws){
  unsigned gid = blockIdx.x * 256u + threadIdx.x;
  if (gid < 57408u) ws[gid] = 0u;   // zero ctr + c + hfrag + hxfer (229632 B)
}

extern "C" __global__ void lstm_pack_kernel(
    const float* __restrict__ Wih0, const float* __restrict__ Whh0,
    const float* __restrict__ Wih1, const float* __restrict__ Whh1,
    const float* __restrict__ bih0, const float* __restrict__ bhh0,
    const float* __restrict__ bih1, const float* __restrict__ bhh1,
    _Float16* __restrict__ Wpack, float* __restrict__ biasCat)
{
  size_t gid = (size_t)blockIdx.x * 256 + threadIdx.x;
  const size_t NW = 2ull * 1024 * 2048;       // 2 layers x K x 4H
  if (gid < NW){
    int l = (int)(gid >> 21);
    unsigned rr = (unsigned)(gid & 2097151u);
    int k = (int)(rr >> 11);                  // 0..1023
    int n = (int)(rr & 2047u);                // 0..2047 (gate column)
    const float* Wih = l ? Wih1 : Wih0;
    const float* Whh = l ? Whh1 : Whh0;
    float v = (k < 512) ? Wih[(size_t)n * 512 + k]
                        : Whh[(size_t)n * 512 + (k - 512)];
    int kt = k >> 5, kk = k & 31, nt = n >> 4, n16 = n & 15;
    int ln = n16 + ((kk >> 3) & 1) * 16;      // B-frag lane
    int ix = (kk & 7) | ((kk >> 4) << 3);     // B-frag half index
    Wpack[(size_t)l * (32 * 128 * 512) + (((kt * 128 + nt) << 9) + ln * 16 + ix)]
        = (_Float16)v;
  } else if (gid < NW + 4096){
    unsigned i = (unsigned)(gid - NW);
    int l = (int)(i >> 11), n = (int)(i & 2047u);
    biasCat[l * 2048 + n] = l ? (bih1[n] + bhh1[n]) : (bih0[n] + bhh0[n]);
  }
}

extern "C" __global__ __launch_bounds__(256, 1)
void lstm_scan_kernel(const float* __restrict__ x,
                      const _Float16* __restrict__ Wpack,
                      const float* __restrict__ biasCat,
                      float* __restrict__ cstate,
                      _Float16* __restrict__ hfrag,
                      _Float16* __restrict__ hxfer,
                      float* __restrict__ gbuf,
                      unsigned* __restrict__ ctr,
                      float* __restrict__ out)
{
  __shared__ _Float16 afrag[2 * 32 * 512];    // A frags: [mt][kt][512], 64 KB

  const int tid   = threadIdx.x;
  const int wgid  = blockIdx.x;
  const int layer = wgid >> 4;                // WGs 0-15 -> layer0, 16-31 -> layer1
  const int lwg   = wgid & 15;
  const int lane  = tid & 31;
  const int wv    = tid >> 5;
  const int nt    = lwg * 8 + wv;             // N-stripe 0..127 (16 gate cols)

  const _Float16* WL  = Wpack + (size_t)layer * (32 * 128 * 512);
  const float*    bL  = biasCat + layer * 2048;
  float*          gL  = gbuf + layer * (256 * 256);
  float*          cL  = cstate + layer * (32 * 512);
  _Float16*       hfL = hfrag + layer * (2 * 16 * 512);

  const v16h* W16 = (const v16h*)WL;
  const v16h* A16 = (const v16h*)afrag;

  unsigned target = 0;
  for (int r = 0; r <= TT; ++r){
    const int  t      = (layer == 0) ? r : (r - 1);
    const bool active = (layer == 0) ? (r < TT) : (r >= 1);

    // -------- stage A = [x_t | h_{t-1}] into LDS, pre-swizzled -------------
    if (active){
      if (layer == 0){
        // x part (kt 0..15) from f32 x[B,T,512], converted to f16
        for (int i = 0; i < 16; ++i){
          int e4 = i * 256 + tid;             // 4096 quads
          int b  = e4 >> 7;
          int k0 = (e4 & 127) << 2;           // multiple of 4 -> one v4h store
          const float4 v = *(const float4*)(x + ((size_t)b * TT + t) * 512 + k0);
          int mt = b >> 4, m16 = b & 15;
          int kt = k0 >> 5, kk = k0 & 31;
          int ln = m16 + ((kk >> 3) & 1) * 16;
          int ix = (kk & 7) | ((kk >> 4) << 3);
          v4h h4 = { (_Float16)v.x, (_Float16)v.y, (_Float16)v.z, (_Float16)v.w };
          *(v4h*)&afrag[((mt * 32 + kt) << 9) + ln * 16 + ix] = h4;
        }
      } else {
        // x part = layer0's h (already A-swizzled in hxfer): async DMA copy
#if HAS_ASYNC_LDS
        for (int i = 0; i < 8; ++i){
          int cu = i * 256 + tid;             // 2048 x 16B chunks (32 KB)
          int inner = cu & 63, fg = cu >> 6;
          int mt = fg >> 4, kt = fg & 15;
          async_copy16((const char*)hxfer + ((size_t)cu << 4),
                       (char*)afrag + (((mt * 32 + kt) << 10) + (inner << 4)));
        }
#else
        const uint2* src = (const uint2*)hxfer;
        for (int i = 0; i < 16; ++i){
          int u = i * 256 + tid;              // 4096 uint2
          int inner = u & 127, fg = u >> 7;   // fg: [mt][kt]
          int mt = fg >> 4, kt = fg & 15;
          ((uint2*)afrag)[((mt * 32 + kt) << 7) + inner] = src[u];
        }
#endif
      }
      // h part (kt 16..31) from this layer's A-swizzled h state
#if HAS_ASYNC_LDS
      for (int i = 0; i < 8; ++i){
        int cu = i * 256 + tid;               // 2048 x 16B chunks (32 KB)
        int inner = cu & 63, fg = cu >> 6;
        int mt = fg >> 4, kt = fg & 15;
        async_copy16((const char*)hfL + ((size_t)cu << 4),
                     (char*)afrag + (((mt * 32 + 16 + kt) << 10) + (inner << 4)));
      }
#else
      {
        const uint2* src = (const uint2*)hfL;
        for (int i = 0; i < 16; ++i){
          int u = i * 256 + tid;
          int inner = u & 127, fg = u >> 7;
          int mt = fg >> 4, kt = fg & 15;
          ((uint2*)afrag)[((mt * 32 + 16 + kt) << 7) + inner] = src[u];
        }
      }
#endif
      async_wait();
    }
    __syncthreads();

    // -------- GEMM: both M-tiles of this wave's N-stripe, K=1024 -----------
    if (active){
      v8f acc0 = {}, acc1 = {};
      for (int kt = 0; kt < 32; ++kt){
        v16h bfr = W16[((kt * 128 + nt) << 5) + lane];
        if (kt + 1 < 32)
          __builtin_prefetch(&W16[(((kt + 1) * 128 + nt) << 5) + lane], 0, 3);
        v16h a0 = A16[(kt << 5) + lane];
        v16h a1 = A16[((32 + kt) << 5) + lane];
        acc0 = __builtin_amdgcn_wmma_f32_16x16x32_f16(
                   false, a0, false, bfr, (short)0, acc0, false, false);
        acc1 = __builtin_amdgcn_wmma_f32_16x16x32_f16(
                   false, a1, false, bfr, (short)0, acc1, false, false);
      }
      *(v8f*)&gL[(nt << 8) + lane * 8]         = acc0;   // tile (mt=0, nt)
      *(v8f*)&gL[((128 + nt) << 8) + lane * 8] = acc1;   // tile (mt=1, nt)
    }

    target += NWG; grid_sync(ctr, target);

    // -------- elementwise LSTM cell update (f32) ---------------------------
    if (active){
      for (int i = 0; i < 4; ++i){
        int e = i * 4096 + lwg * 256 + tid;   // [0,16384): e = b*512 + j
        int j = e & 511, b = e >> 9;
        int mt = b >> 4, m = b & 15;
        int lnrow = ((m >> 3) & 1) * 16;
        int vv = m & 7;
        float g4[4];
        #pragma unroll
        for (int q = 0; q < 4; ++q){          // gather i,f,g,o from C tiles
          int n = q * 512 + j;
          int ntq = n >> 4, n16 = n & 15;
          g4[q] = gL[((mt * 128 + ntq) << 8) + (n16 + lnrow) * 8 + vv] + bL[n];
        }
        float ig  = fsig(g4[0]);
        float fg_ = fsig(g4[1]);
        float gg  = ftanh(g4[2]);
        float og  = fsig(g4[3]);
        float cn  = fg_ * cL[e] + ig * gg;
        cL[e] = cn;
        float h = og * ftanh(cn);
        // write h back in A-fragment swizzle for next step's staging
        int ktj = j >> 5, kk = j & 31;
        int ln2 = m + ((kk >> 3) & 1) * 16;
        int ix2 = (kk & 7) | ((kk >> 4) << 3);
        int off = ((mt * 16 + ktj) << 9) + ln2 * 16 + ix2;
        _Float16 hh = (_Float16)h;
        hfL[off] = hh;
        if (layer == 0) hxfer[off] = hh;                       // feed layer1
        else out[((size_t)b * TT + t) * 512 + j] = h;          // final output
      }
    }

    target += NWG; grid_sync(ctr, target);
  }
}

extern "C" void kernel_launch(void* const* d_in, const int* in_sizes, int n_in,
                              void* d_out, int out_size, void* d_ws, size_t ws_size,
                              hipStream_t stream)
{
  (void)in_sizes; (void)n_in; (void)out_size; (void)ws_size;
  const float* x    = (const float*)d_in[0];
  const float* Wih0 = (const float*)d_in[1];
  const float* bih0 = (const float*)d_in[2];
  const float* Whh0 = (const float*)d_in[3];
  const float* bhh0 = (const float*)d_in[4];
  const float* Wih1 = (const float*)d_in[5];
  const float* bih1 = (const float*)d_in[6];
  const float* Whh1 = (const float*)d_in[7];
  const float* bhh1 = (const float*)d_in[8];
  float* out = (float*)d_out;

  char* w = (char*)d_ws;
  unsigned* ctr   = (unsigned*)(w + 0);
  float*    cst   = (float*)   (w + 256);
  _Float16* hfrag = (_Float16*)(w + 131328);
  _Float16* hxfer = (_Float16*)(w + 196864);
  float*    biasC = (float*)   (w + 229632);
  float*    gb    = (float*)   (w + 246016);
  _Float16* Wpack = (_Float16*)(w + 508160);

  hipLaunchKernelGGL(lstm_init_kernel, dim3(225), dim3(256), 0, stream,
                     (unsigned*)w);
  hipLaunchKernelGGL(lstm_pack_kernel, dim3(16400), dim3(256), 0, stream,
                     Wih0, Whh0, Wih1, Whh1, bih0, bhh0, bih1, bhh1,
                     Wpack, biasC);
  hipLaunchKernelGGL(lstm_scan_kernel, dim3(NWG), dim3(256), 0, stream,
                     x, Wpack, biasC, cst, hfrag, hxfer, gb, ctr, out);
}
